// GlobalLocalAttention_84945863180766
// MI455X (gfx1250) — matrix operand order repaired and checked by reading, not compiled, not run on hardware
//
#include <hip/hip_runtime.h>
#include <hip/hip_bf16.h>

// Problem constants (from reference): B=8, S=4096, H=1024, NH=16, D=64
#define B_  8
#define S_  4096
#define H_  1024
#define NH_ 16
#define D_  64

typedef __attribute__((ext_vector_type(16))) __bf16 v16bf;
typedef __attribute__((ext_vector_type(8)))  float  v8f;

union Frag16 {
    v16bf v;
    uint4 q[2];
};

// ---------------------------------------------------------------------------
// Kernel 1: fused QKV GEMM, bf16 WMMA, f32 accumulate.
//   C[m, n] = sum_k h[m,k] * W[n,k] + bias[n]
//   m in [0, B*S), n in [0, 2048): n<1024 -> Wq/bq -> qws, else Wkv/bkv -> kws
//   Output layout: (B, NH, S, D) row-major for both q and k.
// Block tile 128(M) x 128(N) x 32(K); 256 threads = 8 waves; each wave does a
// 32x64 sub-tile = 2x4 WMMA 16x16 tiles -> 8 v_wmma_f32_16x16x32_bf16 / K-step.
// f32 -> bf16 conversion uses native __bf16 casts so clang can select the
// packed bf16 convert hardware instead of a bit-twiddled integer sequence.
// ---------------------------------------------------------------------------
#define BM 128
#define BN 128
#define BK 32
#define LDK 40   // BK + 8 pad: rows are 80B -> 16B aligned frags, conflict-free half-wave

__global__ __launch_bounds__(256)
void qkv_gemm_wmma(const float* __restrict__ hs,
                   const float* __restrict__ Wq,  const float* __restrict__ bq,
                   const float* __restrict__ Wkv, const float* __restrict__ bkv,
                   float* __restrict__ qws, float* __restrict__ kws)
{
    __shared__ __bf16 As[BM][LDK];   // A tile: [m][k] bf16
    __shared__ __bf16 Bs[BN][LDK];   // B tile stored [n][k] bf16

    const int tid   = threadIdx.x;
    const int lane  = tid & 31;
    const int wave  = tid >> 5;
    const int waveM = wave & 3;    // 4 waves along M
    const int waveN = wave >> 2;   // 2 waves along N

    const int nBlock = blockIdx.x;             // 0..15  (2048 / 128)
    const int mBlock = blockIdx.y;             // 0..255 (32768 / 128)
    const bool isQ   = nBlock < 8;

    const float* __restrict__ W    = isQ ? Wq  : Wkv;
    const float* __restrict__ bias = isQ ? bq  : bkv;
    float*       __restrict__ outp = isQ ? qws : kws;
    const int nbase = (isQ ? nBlock : nBlock - 8) * BN;   // channel base 0..1023
    const int mbase = mBlock * BM;

    v8f acc[2][4] = {};

    for (int k0 = 0; k0 < H_; k0 += BK) {
        // ---- cooperative load: 128x32 f32 each for A and B, convert to bf16 ----
#pragma unroll
        for (int i = 0; i < 4; ++i) {
            const int idx = tid + i * 256;          // 0..1023
            const int row = idx >> 3;               // 0..127
            const int c4  = (idx & 7) << 2;         // 0,4,...,28

            const float4 va = *(const float4*)(hs + (size_t)(mbase + row) * H_ + k0 + c4);
            As[row][c4 + 0] = (__bf16)va.x;
            As[row][c4 + 1] = (__bf16)va.y;
            As[row][c4 + 2] = (__bf16)va.z;
            As[row][c4 + 3] = (__bf16)va.w;

            const float4 vb = *(const float4*)(W + (size_t)(nbase + row) * H_ + k0 + c4);
            Bs[row][c4 + 0] = (__bf16)vb.x;
            Bs[row][c4 + 1] = (__bf16)vb.y;
            Bs[row][c4 + 2] = (__bf16)vb.z;
            Bs[row][c4 + 3] = (__bf16)vb.w;

            if (k0 + BK < H_) {   // hint next K tile -> global_prefetch_b8
                __builtin_prefetch(hs + (size_t)(mbase + row) * H_ + k0 + BK + c4, 0, 1);
                __builtin_prefetch(W  + (size_t)(nbase + row) * H_ + k0 + BK + c4, 0, 1);
            }
        }
        __syncthreads();

        // ---- build fragments per ISA 7.12.2 layouts ----
        // A (16-bit, 16x32): lane<16 -> M=lane,    K = {0..7, 16..23}
        //                    lane>=16 -> M=lane-16, K = {8..15, 24..31}
        const int kA   = (lane < 16) ? 0 : 8;
        // B (32x16, from [n][k] LDS): lane<16 -> K = {0..7, 16..23}, lane>=16 -> K = {8..15, 24..31}; N=lane%16
        const int kB   = (lane < 16) ? 0 : 8;
        const int rsel = lane & 15;

        Frag16 aF[2], bF[4];
#pragma unroll
        for (int i = 0; i < 2; ++i) {
            const int m = waveM * 32 + i * 16 + rsel;
            aF[i].q[0] = *(const uint4*)&As[m][kA];
            aF[i].q[1] = *(const uint4*)&As[m][kA + 16];
        }
#pragma unroll
        for (int j = 0; j < 4; ++j) {
            const int n = waveN * 64 + j * 16 + rsel;
            bF[j].q[0] = *(const uint4*)&Bs[n][kB];
            bF[j].q[1] = *(const uint4*)&Bs[n][kB + 16];
        }

#pragma unroll
        for (int i = 0; i < 2; ++i)
#pragma unroll
            for (int j = 0; j < 4; ++j)
                acc[i][j] = __builtin_amdgcn_wmma_f32_16x16x32_bf16(
                    false, aF[i].v, false, bF[j].v, (short)0, acc[i][j], false, false);

        __syncthreads();
    }

    // ---- epilogue: bias add, scatter to (B, NH, S, D) layout ----
    const int ncol = lane & 15;
    const int mofs = (lane < 16) ? 0 : 8;
#pragma unroll
    for (int j = 0; j < 4; ++j) {
        const int ch = nbase + waveN * 64 + j * 16 + ncol;   // channel 0..1023
        const float bv = bias[ch];
        const int hh = ch >> 6;
        const int dd = ch & 63;
#pragma unroll
        for (int i = 0; i < 2; ++i) {
            const int mrow0 = mbase + waveM * 32 + i * 16 + mofs;
#pragma unroll
            for (int v = 0; v < 8; ++v) {
                const int m = mrow0 + v;         // global position p = b*S + s
                const int b = m >> 12;           // / 4096
                const int s = m & 4095;
                const size_t o = (((size_t)(b * NH_ + hh)) * S_ + s) * D_ + dd;
                outp[o] = acc[i][j][v] + bv;
            }
        }
    }
}

// ---------------------------------------------------------------------------
// Kernel 2: global attention. One block per (b,h).
//   scores[s] = scale * <q[b,h,0,:], k[b,h,s,:]> + mask[b,s]; softmax over s;
//   g_vec[b,h,:] = sum_s attn[s] * k[b,h,s,:]
// ---------------------------------------------------------------------------
__global__ __launch_bounds__(256)
void global_attn(const float* __restrict__ qws, const float* __restrict__ kws,
                 const float* __restrict__ mask, float* __restrict__ gvec)
{
    __shared__ float sc[S_];      // 16 KB
    __shared__ float red[256];
    __shared__ float gq[D_];

    const int bh   = blockIdx.x;        // b*NH + h
    const int b    = bh >> 4;
    const int tid  = threadIdx.x;
    const int lane = tid & 31;
    const int wave = tid >> 5;
    const float scale = 0.125f;         // 1/sqrt(64)

    if (tid < D_) gq[tid] = qws[((size_t)bh * S_) * D_ + tid];
    __syncthreads();

    // scores: one wave per key row, lane-split dot + shuffle reduce
    const float g0 = gq[lane], g1 = gq[lane + 32];
    for (int s = wave; s < S_; s += 8) {
        const float* kr = kws + ((size_t)bh * S_ + s) * D_;
        float p = g0 * kr[lane] + g1 * kr[lane + 32];
#pragma unroll
        for (int off = 16; off >= 1; off >>= 1) p += __shfl_xor(p, off, 32);
        if (lane == 0) sc[s] = p * scale + mask[(size_t)b * S_ + s];
    }
    __syncthreads();

    // block max
    float mx = -3.4e38f;
    for (int s = tid; s < S_; s += 256) mx = fmaxf(mx, sc[s]);
    red[tid] = mx;
    __syncthreads();
#pragma unroll
    for (int w = 128; w >= 1; w >>= 1) {
        if (tid < w) red[tid] = fmaxf(red[tid], red[tid + w]);
        __syncthreads();
    }
    mx = red[0];
    __syncthreads();

    // exponentiate in place + block sum
    float sm = 0.f;
    for (int s = tid; s < S_; s += 256) {
        const float e = __expf(sc[s] - mx);
        sc[s] = e;
        sm += e;
    }
    red[tid] = sm;
    __syncthreads();
#pragma unroll
    for (int w = 128; w >= 1; w >>= 1) {
        if (tid < w) red[tid] += red[tid + w];
        __syncthreads();
    }
    const float inv = 1.0f / red[0];
    __syncthreads();

    // weighted sum of keys, d-parallel, 4 s-groups
    const int d   = tid & 63;
    const int grp = tid >> 6;
    float acc = 0.f;
    for (int s = grp; s < S_; s += 4)
        acc += sc[s] * kws[((size_t)bh * S_ + s) * D_ + d];
    red[tid] = acc;
    __syncthreads();
    if (tid < D_)
        gvec[bh * D_ + tid] =
            (red[tid] + red[tid + 64] + red[tid + 128] + red[tid + 192]) * inv;
}

// ---------------------------------------------------------------------------
// Kernel 3: local 2-way softmax + blend. One wave per (b,h,s).
// ---------------------------------------------------------------------------
__global__ __launch_bounds__(256)
void local_attn(const float* __restrict__ qws, const float* __restrict__ kws,
                const float* __restrict__ gvec, float* __restrict__ ctx,
                float* __restrict__ lattn)
{
    const int tid  = threadIdx.x;
    const int lane = tid & 31;
    const int wave = tid >> 5;
    const long pos = (long)blockIdx.x * 8 + wave;   // 0 .. B*NH*S-1
    const int bh = (int)(pos >> 12);
    const int s  = (int)(pos & 4095);
    const int b  = bh >> 4;
    const int hh = bh & 15;

    const size_t base = (size_t)pos * D_;
    const float q0 = qws[base + lane],      q1 = qws[base + lane + 32];
    const float k0 = kws[base + lane],      k1 = kws[base + lane + 32];
    const float g0 = gvec[bh * D_ + lane],  g1 = gvec[bh * D_ + lane + 32];

    const size_t co = ((size_t)(b * S_ + s)) * H_ + hh * D_;
    if (s == 0) {
        ctx[co + lane]      = g0;
        ctx[co + lane + 32] = g1;
    } else {
        float ps = q0 * k0 + q1 * k1;
        float pg = q0 * g0 + q1 * g1;
#pragma unroll
        for (int off = 16; off >= 1; off >>= 1) {
            ps += __shfl_xor(ps, off, 32);
            pg += __shfl_xor(pg, off, 32);
        }
        const float ss = ps * 0.125f, sg = pg * 0.125f;
        const float m  = fmaxf(ss, sg);
        const float e0 = __expf(ss - m), e1 = __expf(sg - m);
        const float inv = 1.0f / (e0 + e1);
        const float a0 = e0 * inv, a1 = e1 * inv;
        ctx[co + lane]      = a0 * k0 + a1 * g0;
        ctx[co + lane + 32] = a0 * k1 + a1 * g1;
        if (lane == 0) {
            const size_t lo = ((size_t)bh * (S_ - 1) + (s - 1)) * 2;
            lattn[lo]     = a0;
            lattn[lo + 1] = a1;
        }
    }
}

// ---------------------------------------------------------------------------
// Launch. Workspace: q (B*S*H f32) + k (B*S*H f32) + gvec (B*NH*D f32)
//                  ~= 268.5 MB.
// Output: ctx (B*S*H f32) followed by local_attentions (B*NH*(S-1)*2 f32).
// ---------------------------------------------------------------------------
extern "C" void kernel_launch(void* const* d_in, const int* in_sizes, int n_in,
                              void* d_out, int out_size, void* d_ws, size_t ws_size,
                              hipStream_t stream)
{
    const float* hs   = (const float*)d_in[0];
    const float* mask = (const float*)d_in[1];
    const float* Wq   = (const float*)d_in[2];
    const float* bq   = (const float*)d_in[3];
    const float* Wkv  = (const float*)d_in[4];
    const float* bkv  = (const float*)d_in[5];

    float* out  = (float*)d_out;
    float* qws  = (float*)d_ws;
    float* kws  = qws + (size_t)B_ * S_ * H_;
    float* gvec = kws + (size_t)B_ * S_ * H_;

    float* ctx   = out;
    float* lattn = out + (size_t)B_ * S_ * H_;

    dim3 g1(2 * H_ / BN, (B_ * S_) / BM);           // (16, 256)
    qkv_gemm_wmma<<<g1, 256, 0, stream>>>(hs, Wq, bq, Wkv, bkv, qws, kws);

    global_attn<<<B_ * NH_, 256, 0, stream>>>(qws, kws, mask, gvec);

    local_attn<<<(B_ * NH_ * S_) / 8, 256, 0, stream>>>(qws, kws, gvec, ctx, lattn);
}
